// RemainEatingTimeBackbone_32152125178244
// MI455X (gfx1250) — matrix operand order repaired and checked by reading, not compiled, not run on hardware
//
#include <hip/hip_runtime.h>
#include <hip/hip_bf16.h>
#include <stdint.h>

#define B_   8
#define L_   512
#define D_   512
#define H_   8
#define HD_  64
#define M_   (B_*L_)      // 4096 tokens
#define FF_  2048
#define PADV 101

typedef __attribute__((ext_vector_type(16))) _Float16 v16h;
typedef __attribute__((ext_vector_type(8)))  float    v8f;

// ======================= small elementwise kernels =======================

__global__ __launch_bounds__(256) void embed_kernel(
    const int* __restrict__ ids, const float* __restrict__ emb,
    const float* __restrict__ pe, float* __restrict__ out)
{
  int t  = blockIdx.x;        // token 0..M_-1
  int l  = t & (L_ - 1);
  int id = ids[t];
  int tid = threadIdx.x;
  size_t o = (size_t)t * D_;
  out[o + tid]       = emb[(size_t)id * D_ + tid]       + pe[(size_t)l * D_ + tid];
  out[o + tid + 256] = emb[(size_t)id * D_ + tid + 256] + pe[(size_t)l * D_ + tid + 256];
}

// LayerNorm over D=512, f32 in -> f16 out (feeds WMMA GEMMs)
__global__ __launch_bounds__(256) void ln_kernel(
    const float* __restrict__ x, const float* __restrict__ w,
    const float* __restrict__ b, _Float16* __restrict__ out)
{
  __shared__ float s1[256], s2[256];
  int t = blockIdx.x, tid = threadIdx.x;
  const float* row = x + (size_t)t * D_;
  float v0 = row[tid], v1 = row[tid + 256];
  s1[tid] = v0 + v1;
  s2[tid] = v0 * v0 + v1 * v1;
  __syncthreads();
  for (int o = 128; o > 0; o >>= 1) {
    if (tid < o) { s1[tid] += s1[tid + o]; s2[tid] += s2[tid + o]; }
    __syncthreads();
  }
  float mu  = s1[0] * (1.0f / D_);
  float var = s2[0] * (1.0f / D_) - mu * mu;
  float inv = rsqrtf(var + 1e-5f);
  _Float16* orow = out + (size_t)t * D_;
  orow[tid]       = (_Float16)((v0 - mu) * inv * w[tid]       + b[tid]);
  orow[tid + 256] = (_Float16)((v1 - mu) * inv * w[tid + 256] + b[tid + 256]);
}

// Wt[n][k] = (f16) W[k][n]   (write-coalesced transpose+downconvert)
__global__ __launch_bounds__(256) void convt_kernel(
    const float* __restrict__ W, _Float16* __restrict__ Wt, int K, int N)
{
  size_t i = (size_t)blockIdx.x * 256 + threadIdx.x;
  if (i >= (size_t)K * N) return;
  int k = (int)(i % K), n = (int)(i / K);
  Wt[i] = (_Float16)W[(size_t)k * N + n];
}

// masked softmax over keys; scores f32 [B*H][L][L] -> probs f16, mask = pad | causal
__global__ __launch_bounds__(256) void softmax_kernel(
    const float* __restrict__ scores, _Float16* __restrict__ probs,
    const int* __restrict__ ids, int causal)
{
  __shared__ float red[256];
  int q = blockIdx.x, bh = blockIdx.y, b = bh >> 3;   // H_ == 8
  int tid = threadIdx.x;
  const float* row  = scores + ((size_t)bh * L_ + q) * L_;
  _Float16*    prow = probs  + ((size_t)bh * L_ + q) * L_;
  int k0 = tid, k1 = tid + 256;
  float s0 = row[k0], s1v = row[k1];
  if (ids[b * L_ + k0] == PADV || (causal && k0 > q)) s0  = -1e30f;
  if (ids[b * L_ + k1] == PADV || (causal && k1 > q)) s1v = -1e30f;
  red[tid] = fmaxf(s0, s1v);
  __syncthreads();
  for (int o = 128; o > 0; o >>= 1) { if (tid < o) red[tid] = fmaxf(red[tid], red[tid + o]); __syncthreads(); }
  float mx = red[0];
  __syncthreads();
  float e0 = __expf(s0 - mx), e1 = __expf(s1v - mx);
  red[tid] = e0 + e1;
  __syncthreads();
  for (int o = 128; o > 0; o >>= 1) { if (tid < o) red[tid] += red[tid + o]; __syncthreads(); }
  float inv = 1.0f / red[0];
  prow[k0] = (_Float16)(e0 * inv);
  prow[k1] = (_Float16)(e1 * inv);
}

// ======================= WMMA GEMM =======================
// C[M][N] = alpha * A[M][K] * Bt[N][K]^T + bias[N] (+ res)
// 256 threads = 8 waves as 4(M)x2(N); each wave owns a 32x32 register tile
// (2x2 grid of 16x16 WMMA tiles) -> 4 v_wmma per 8 b128 loads per K-step.
// Block tile = 128(M) x 64(N); K step 32. All model dims divide exactly.
// Fragment layouts per CDNA5 ISA 7.12.2 (wave32):
//   A lane l: m=l&15, k = i + 8*(l>>4) + (i>=8 ? 8 : 0)  -> two contiguous 16B chunks
//   B lane l: n=l&15, k = i + 8*(l>>4) + (i>=8 ? 8 : 0)  -> two contiguous 16B chunks of Bt row n
//   C lane l: n=l&15, m = j + 8*(l>>4)

enum { EPI_F32_RES = 0, EPI_F16 = 1, EPI_F16_RELU = 2, EPI_F16_TRANS = 3, EPI_F32_SCALE = 4 };

union Frag  { uint4 u[2]; v16h h; };
union St8h  { uint4 u; _Float16 h[8]; };

template<int MODE>
__global__ __launch_bounds__(256) void gemm_wmma(
    const _Float16* __restrict__ A, const _Float16* __restrict__ Bt,
    const float* __restrict__ bias, const float* res, void* outp,
    int Ksz, int lda, int ldb, int ldc, int ldres,
    int zdiv, long long sA1, long long sA2, long long sB1, long long sB2,
    long long sC1, long long sC2, float alpha)
{
  int w    = threadIdx.x >> 5;
  int lane = threadIdx.x & 31;
  int half = lane >> 4, l15 = lane & 15;
  int tileM = blockIdx.y * 128 + (w >> 1) * 32;
  int tileN = blockIdx.x * 64  + (w & 1)  * 32;
  long long z1 = blockIdx.z / zdiv, z2 = blockIdx.z % zdiv;

  const _Float16* arow0 = A  + z1 * sA1 + z2 * sA2 + (size_t)(tileM + l15) * lda;
  const _Float16* arow1 = arow0 + (size_t)16 * lda;
  const _Float16* brow0 = Bt + z1 * sB1 + z2 * sB2 + (size_t)(tileN + l15) * ldb;
  const _Float16* brow1 = brow0 + (size_t)16 * ldb;
  long long offC = z1 * sC1 + z2 * sC2;

  v8f acc[2][2] = {{{}, {}}, {{}, {}}};
  for (int k0 = 0; k0 < Ksz; k0 += 32) {
    Frag a0, a1, b0, b1;
    a0.u[0] = *(const uint4*)(arow0 + k0 + 8 * half);
    a0.u[1] = *(const uint4*)(arow0 + k0 + 16 + 8 * half);
    a1.u[0] = *(const uint4*)(arow1 + k0 + 8 * half);
    a1.u[1] = *(const uint4*)(arow1 + k0 + 16 + 8 * half);
    b0.u[0] = *(const uint4*)(brow0 + k0 + 8 * half);
    b0.u[1] = *(const uint4*)(brow0 + k0 + 16 + 8 * half);
    b1.u[0] = *(const uint4*)(brow1 + k0 + 8 * half);
    b1.u[1] = *(const uint4*)(brow1 + k0 + 16 + 8 * half);
    acc[0][0] = __builtin_amdgcn_wmma_f32_16x16x32_f16(false, a0.h, false, b0.h, (short)0, acc[0][0], false, false);
    acc[0][1] = __builtin_amdgcn_wmma_f32_16x16x32_f16(false, a0.h, false, b1.h, (short)0, acc[0][1], false, false);
    acc[1][0] = __builtin_amdgcn_wmma_f32_16x16x32_f16(false, a1.h, false, b0.h, (short)0, acc[1][0], false, false);
    acc[1][1] = __builtin_amdgcn_wmma_f32_16x16x32_f16(false, a1.h, false, b1.h, (short)0, acc[1][1], false, false);
  }

  float bv[2];
  bv[0] = bias ? bias[tileN + l15]      : 0.0f;
  bv[1] = bias ? bias[tileN + 16 + l15] : 0.0f;

  #pragma unroll
  for (int im = 0; im < 2; ++im) {
    #pragma unroll
    for (int jn = 0; jn < 2; ++jn) {
      int tm = tileM + 16 * im;
      int tn = tileN + 16 * jn;
      v8f a = acc[im][jn];
      float bb = bv[jn];

      if constexpr (MODE == EPI_F16_TRANS) {
        St8h st;
        #pragma unroll
        for (int j = 0; j < 8; ++j) st.h[j] = (_Float16)(a[j] * alpha + bb);
        _Float16* o = (_Float16*)outp + offC + (size_t)(tn + l15) * ldc + tm + 8 * half;
        *(uint4*)o = st.u;                       // contiguous 16B transposed store
      } else if constexpr (MODE == EPI_F32_RES) {
        float* o = (float*)outp + offC;
        #pragma unroll
        for (int j = 0; j < 8; ++j) {
          int mg = tm + 8 * half + j, ng = tn + l15;
          o[(size_t)mg * ldc + ng] = a[j] * alpha + bb + res[(size_t)mg * ldres + ng];
        }
      } else if constexpr (MODE == EPI_F32_SCALE) {
        float* o = (float*)outp + offC;
        #pragma unroll
        for (int j = 0; j < 8; ++j)
          o[(size_t)(tm + 8 * half + j) * ldc + tn + l15] = a[j] * alpha + bb;
      } else {  // EPI_F16 / EPI_F16_RELU
        _Float16* o = (_Float16*)outp + offC;
        #pragma unroll
        for (int j = 0; j < 8; ++j) {
          float v = a[j] * alpha + bb;
          if constexpr (MODE == EPI_F16_RELU) v = fmaxf(v, 0.0f);
          o[(size_t)(tm + 8 * half + j) * ldc + tn + l15] = (_Float16)v;
        }
      }
    }
  }
}

// ======================= host orchestration =======================

struct AttnP { const float *lnw,*lnb,*qW,*qb,*kW,*kb,*vW,*vb,*oW,*ob; };
struct FpnP  { const float *lnw,*lnb,*f1W,*f1b,*f2W,*f2b; };

struct WS {
  float *x, *y, *scores;
  _Float16 *ln16, *lnkv16, *q16, *k16, *vT16, *ao16, *hid16, *probs, *wt;
};

static void gemm(hipStream_t st, int mode, const _Float16* A, const _Float16* Bt,
                 const float* bias, const float* res, void* out,
                 int Msz, int Nsz, int Ksz, int lda, int ldb, int ldc, int ldres,
                 int batch, int zdiv,
                 long long sA1, long long sA2, long long sB1, long long sB2,
                 long long sC1, long long sC2, float alpha)
{
  dim3 g(Nsz / 64, Msz / 128, batch), blk(256, 1, 1);
  switch (mode) {
    case EPI_F32_RES:   gemm_wmma<EPI_F32_RES>  <<<g, blk, 0, st>>>(A, Bt, bias, res, out, Ksz, lda, ldb, ldc, ldres, zdiv, sA1, sA2, sB1, sB2, sC1, sC2, alpha); break;
    case EPI_F16:       gemm_wmma<EPI_F16>      <<<g, blk, 0, st>>>(A, Bt, bias, res, out, Ksz, lda, ldb, ldc, ldres, zdiv, sA1, sA2, sB1, sB2, sC1, sC2, alpha); break;
    case EPI_F16_RELU:  gemm_wmma<EPI_F16_RELU> <<<g, blk, 0, st>>>(A, Bt, bias, res, out, Ksz, lda, ldb, ldc, ldres, zdiv, sA1, sA2, sB1, sB2, sC1, sC2, alpha); break;
    case EPI_F16_TRANS: gemm_wmma<EPI_F16_TRANS><<<g, blk, 0, st>>>(A, Bt, bias, res, out, Ksz, lda, ldb, ldc, ldres, zdiv, sA1, sA2, sB1, sB2, sC1, sC2, alpha); break;
    default:            gemm_wmma<EPI_F32_SCALE><<<g, blk, 0, st>>>(A, Bt, bias, res, out, Ksz, lda, ldb, ldc, ldres, zdiv, sA1, sA2, sB1, sB2, sC1, sC2, alpha); break;
  }
}

static void gemm1(hipStream_t st, int mode, const _Float16* A, const _Float16* Bt,
                  const float* bias, const float* res, void* out,
                  int Msz, int Nsz, int Ksz, int lda, int ldb, int ldc, int ldres,
                  float alpha)
{
  gemm(st, mode, A, Bt, bias, res, out, Msz, Nsz, Ksz, lda, ldb, ldc, ldres,
       1, 1, 0, 0, 0, 0, 0, 0, alpha);
}

static void convW(hipStream_t st, const float* W, _Float16* Wt, int K, int N)
{
  size_t tot = (size_t)K * N;
  convt_kernel<<<dim3((unsigned)((tot + 255) / 256)), dim3(256), 0, st>>>(W, Wt, K, N);
}

static void run_attn(hipStream_t st, WS& w, float* qbuf, float* kvbuf,
                     const AttnP& p, const int* ids, int causal)
{
  ln_kernel<<<M_, 256, 0, st>>>(qbuf, p.lnw, p.lnb, w.ln16);
  _Float16* lnkv = w.ln16;
  if (kvbuf != qbuf) {
    ln_kernel<<<M_, 256, 0, st>>>(kvbuf, p.lnw, p.lnb, w.lnkv16);
    lnkv = w.lnkv16;
  }
  // Q, K projections (f16, row-major); V projection stored transposed [D][M]
  convW(st, p.qW, w.wt, D_, D_);
  gemm1(st, EPI_F16, w.ln16, w.wt, p.qb, nullptr, w.q16, M_, D_, D_, D_, D_, D_, 0, 1.0f);
  convW(st, p.kW, w.wt, D_, D_);
  gemm1(st, EPI_F16, lnkv, w.wt, p.kb, nullptr, w.k16, M_, D_, D_, D_, D_, D_, 0, 1.0f);
  convW(st, p.vW, w.wt, D_, D_);
  gemm1(st, EPI_F16_TRANS, lnkv, w.wt, p.vb, nullptr, w.vT16, M_, D_, D_, D_, D_, M_, 0, 1.0f);

  // scores[b][h][q][k] = (Q . K^T)/8, batched over z=b*H+h
  gemm(st, EPI_F32_SCALE, w.q16, w.k16, nullptr, nullptr, w.scores,
       L_, L_, HD_, D_, D_, L_, 0,
       B_ * H_, H_,
       (long long)L_ * D_, (long long)HD_,
       (long long)L_ * D_, (long long)HD_,
       (long long)H_ * L_ * L_, (long long)L_ * L_,
       0.125f);
  softmax_kernel<<<dim3(L_, B_ * H_), 256, 0, st>>>(w.scores, w.probs, ids, causal);

  // attn_out[b*L+q][h*64+d] = P . V, batched over z=b*H+h; B-operand is vT (contiguous K)
  gemm(st, EPI_F16, w.probs, w.vT16, nullptr, nullptr, w.ao16,
       L_, HD_, L_, L_, M_, D_, 0,
       B_ * H_, H_,
       (long long)H_ * L_ * L_, (long long)L_ * L_,
       (long long)L_, (long long)HD_ * M_,
       (long long)L_ * D_, (long long)HD_,
       1.0f);

  // output projection + residual (f32, in-place on qbuf)
  convW(st, p.oW, w.wt, D_, D_);
  gemm1(st, EPI_F32_RES, w.ao16, w.wt, p.ob, qbuf, qbuf, M_, D_, D_, D_, D_, D_, D_, 1.0f);
}

static void run_fpn(hipStream_t st, WS& w, float* buf, const FpnP& p)
{
  ln_kernel<<<M_, 256, 0, st>>>(buf, p.lnw, p.lnb, w.ln16);
  convW(st, p.f1W, w.wt, D_, FF_);
  gemm1(st, EPI_F16_RELU, w.ln16, w.wt, p.f1b, nullptr, w.hid16, M_, FF_, D_, D_, D_, FF_, 0, 1.0f);
  convW(st, p.f2W, w.wt, FF_, D_);
  gemm1(st, EPI_F32_RES, w.hid16, w.wt, p.f2b, buf, buf, M_, D_, FF_, FF_, FF_, D_, D_, 1.0f);
}

extern "C" void kernel_launch(void* const* d_in, const int* in_sizes, int n_in,
                              void* d_out, int out_size, void* d_ws, size_t ws_size,
                              hipStream_t stream)
{
  if (n_in < 257) return;
  auto f = [&](int i) { return (const float*)d_in[i]; };

  // Leaf order: sorted jax-pytree order => leaf 0 is dec[0].cross.k.W (262144 elems);
  // raw insertion order => leaf 0 is `remain` (4096 int elems).
  bool sorted = (in_sizes[0] != B_ * L_);

  AttnP encA[6], decS[6], decC[6];
  FpnP  encF[6], decF[6];
  const float *pe, *remEmb, *timEmb;
  const int *remain, *remainTime;

  auto getAttn = [&](int b) {
    AttnP a;
    if (sorted) { a.kW=f(b+0); a.kb=f(b+1); a.lnb=f(b+2); a.lnw=f(b+3);
                  a.oW=f(b+4); a.ob=f(b+5); a.qW=f(b+6); a.qb=f(b+7);
                  a.vW=f(b+8); a.vb=f(b+9); }
    else        { a.lnw=f(b+0); a.lnb=f(b+1); a.qW=f(b+2); a.qb=f(b+3);
                  a.kW=f(b+4); a.kb=f(b+5); a.vW=f(b+6); a.vb=f(b+7);
                  a.oW=f(b+8); a.ob=f(b+9); }
    return a;
  };
  auto getFpn = [&](int b) {
    FpnP q;
    if (sorted) { q.f1W=f(b+0); q.f1b=f(b+1); q.f2W=f(b+2); q.f2b=f(b+3);
                  q.lnb=f(b+4); q.lnw=f(b+5); }
    else        { q.lnw=f(b+0); q.lnb=f(b+1); q.f1W=f(b+2); q.f1b=f(b+3);
                  q.f2W=f(b+4); q.f2b=f(b+5); }
    return q;
  };

  if (sorted) {
    for (int i = 0; i < 6; i++) { int b = i * 26;       decC[i]=getAttn(b); decF[i]=getFpn(b+10); decS[i]=getAttn(b+16); }
    for (int i = 0; i < 6; i++) { int b = 156 + i * 16; encA[i]=getAttn(b); encF[i]=getFpn(b+10); }
    pe = f(252); remEmb = f(253); timEmb = f(254);
    remain = (const int*)d_in[255]; remainTime = (const int*)d_in[256];
  } else {
    remain = (const int*)d_in[0]; remainTime = (const int*)d_in[1];
    remEmb = f(2); timEmb = f(3); pe = f(4);
    for (int i = 0; i < 6; i++) { int b = 5 + i * 16;      encA[i]=getAttn(b); encF[i]=getFpn(b+10); }
    for (int i = 0; i < 6; i++) { int b = 101 + i * 26;    decS[i]=getAttn(b); decC[i]=getAttn(b+10); decF[i]=getFpn(b+20); }
  }

  // ---- workspace carve (~154 MB) ----
  char* base = (char*)d_ws;
  size_t off = 0;
  auto carve = [&](size_t bytes) { void* p = base + off; off = (off + bytes + 255) & ~(size_t)255; return p; };
  WS w;
  w.x      = (float*)carve((size_t)M_ * D_ * 4);
  w.y      = (float*)carve((size_t)M_ * D_ * 4);
  w.scores = (float*)carve((size_t)B_ * H_ * L_ * L_ * 4);
  w.probs  = (_Float16*)carve((size_t)B_ * H_ * L_ * L_ * 2);
  w.ln16   = (_Float16*)carve((size_t)M_ * D_ * 2);
  w.lnkv16 = (_Float16*)carve((size_t)M_ * D_ * 2);
  w.q16    = (_Float16*)carve((size_t)M_ * D_ * 2);
  w.k16    = (_Float16*)carve((size_t)M_ * D_ * 2);
  w.vT16   = (_Float16*)carve((size_t)D_ * M_ * 2);
  w.ao16   = (_Float16*)carve((size_t)M_ * D_ * 2);
  w.hid16  = (_Float16*)carve((size_t)M_ * FF_ * 2);
  w.wt     = (_Float16*)carve((size_t)FF_ * D_ * 2);
  (void)ws_size; (void)out_size;

  // ---- forward ----
  embed_kernel<<<M_, 256, 0, stream>>>(remain,     remEmb, pe, w.x);
  embed_kernel<<<M_, 256, 0, stream>>>(remainTime, timEmb, pe, w.y);

  for (int i = 0; i < 6; i++) {
    run_attn(stream, w, w.x, w.x, encA[i], remain, 0);
    run_fpn(stream, w, w.x, encF[i]);
  }
  for (int i = 0; i < 6; i++) {
    run_attn(stream, w, w.y, w.y, decS[i], remainTime, 1);
    run_attn(stream, w, w.y, w.x, decC[i], remain, 0);
    run_fpn(stream, w, w.y, decF[i]);
  }

  hipMemcpyAsync(d_out, w.y, (size_t)M_ * D_ * sizeof(float),
                 hipMemcpyDeviceToDevice, stream);
}
